// LSTM_MOL_81097572483413
// MI455X (gfx1250) — matrix operand order repaired and checked by reading, not compiled, not run on hardware
//
#include <hip/hip_runtime.h>
#include <hip/hip_bf16.h>

// ---------------- problem constants ----------------
#define T_STEPS 256
#define BATCH   512
#define HID     512
#define G4      2048        // 4*HID gate columns
#define ROWS    16          // batch rows per workgroup == WMMA M
#define BLOCK   512         // 16 waves of 32
#define HPAD    520         // HID + 8 bf16 pad (keeps 16B alignment, staggers banks)

typedef __attribute__((ext_vector_type(16))) __bf16 v16bf;
typedef __attribute__((ext_vector_type(8)))  __bf16 v8bf;
typedef __attribute__((ext_vector_type(8)))  float  v8f;

static __device__ __forceinline__ __bf16 f2bf(float f) {
    unsigned u = __builtin_bit_cast(unsigned, f);
    u += 0x7fffu + ((u >> 16) & 1u);            // round-to-nearest-even
    unsigned short s = (unsigned short)(u >> 16);
    return __builtin_bit_cast(__bf16, s);
}
static __device__ __forceinline__ float sigf(float x) {
    x = fminf(fmaxf(x, -30.0f), 30.0f);
    return 1.0f / (1.0f + __expf(-x));
}
static __device__ __forceinline__ float tanh_fast(float x) {
    x = fminf(fmaxf(x, -15.0f), 15.0f);
    float e = __expf(2.0f * x);
    return (e - 1.0f) / (e + 1.0f);
}

// ---------------- W_hh f32 -> bf16 (row-major (4H,H): K-contiguous per column) ----
__global__ __launch_bounds__(256) void convert_w_kernel(const float* __restrict__ W,
                                                        __bf16* __restrict__ Wb, int n) {
    int i = blockIdx.x * blockDim.x + threadIdx.x;
    if (i < n) Wb[i] = f2bf(W[i]);
}

// ---------------- persistent LSTM scan: one WG owns 16 batch rows for all 256 steps
__global__ __launch_bounds__(BLOCK) void lstm_scan_kernel(
    const float*  __restrict__ x,       // (B, T) row-major
    const float*  __restrict__ w_ih,    // (4H)
    const float*  __restrict__ b_ih,    // (4H)
    const float*  __restrict__ b_hh,    // (4H)
    const __bf16* __restrict__ Wb,      // (4H, H) bf16
    float*        __restrict__ h_out)   // (B, H) final hidden
{
    __shared__ __align__(32) __bf16 h_lds[2][ROWS][HPAD];   // double-buffered h (bf16)
    __shared__ float x_lds[ROWS][T_STEPS];                  // staged inputs

    const int tid     = threadIdx.x;
    const int wave    = tid >> 5;        // 0..15, owns 32 cols of each gate
    const int lane    = tid & 31;
    const int ln      = lane & 15;       // column within 16x16 tile / A row
    const int lhalf   = lane >> 4;       // selects K sub-range of A/B fragments
    const int row0    = blockIdx.x * ROWS;

    // stage x tile (coalesced over t) and zero h buffer 0
    for (int i = tid; i < ROWS * T_STEPS; i += BLOCK) {
        int m = i / T_STEPS, t = i % T_STEPS;
        x_lds[m][t] = x[(size_t)(row0 + m) * T_STEPS + t];
    }
    for (int i = tid; i < ROWS * HPAD; i += BLOCK)
        h_lds[0][i / HPAD][i % HPAD] = f2bf(0.0f);

    // per-lane constants: column j = wave*32 + q*16 + ln, gate g at n = g*H + j
    float wih_r[2][4], bias_r[2][4];
    const __bf16* wptr[2][4];
#pragma unroll
    for (int q = 0; q < 2; ++q)
#pragma unroll
        for (int g = 0; g < 4; ++g) {
            int n = g * HID + wave * 32 + q * 16 + ln;
            wih_r[q][g]  = w_ih[n];
            bias_r[q][g] = b_ih[n] + b_hh[n];
            wptr[q][g]   = Wb + (size_t)n * HID + lhalf * 16;   // B frag: 16 contiguous K
        }

    float c_reg[2][8] = {};      // cell state fragments (D-layout)
    // Opaque zero offset: redefined by empty asm each kk iteration so LICM cannot
    // hoist the W loads out of the time loop (which previously spilled 4KB/lane
    // of hoisted fragments to scratch).
    unsigned wo = 0;

    __syncthreads();

    int cur = 0;
    for (int t = 0; t < T_STEPS; ++t) {
        const int nxt = cur ^ 1;
#pragma unroll
        for (int q = 0; q < 2; ++q) {
            v8f acc0 = {}, acc1 = {}, acc2 = {}, acc3 = {};
#pragma unroll
            for (int kk = 0; kk < HID / 32; ++kk) {
                const int kb = kk * 32;
                asm volatile("" : "+v"(wo));   // defeat LICM of the 4 W loads below
                // A fragment (shared 16 rows of h) from LDS, contiguous K: k = 16*lhalf + s
                const __bf16* hp = &h_lds[cur][ln][kb + lhalf * 16];
                v8bf alo = *(const v8bf*)hp;
                v8bf ahi = *(const v8bf*)(hp + 8);
                union { v16bf v; v8bf h2[2]; } ua;
                ua.h2[0] = alo; ua.h2[1] = ahi;
                const v16bf a = ua.v;
                // B fragments: W streamed from WGP$/L2 (2 MB, fully L2-resident)
                const v16bf bv0 = *(const v16bf*)(wptr[q][0] + kb + wo);
                const v16bf bv1 = *(const v16bf*)(wptr[q][1] + kb + wo);
                const v16bf bv2 = *(const v16bf*)(wptr[q][2] + kb + wo);
                const v16bf bv3 = *(const v16bf*)(wptr[q][3] + kb + wo);
                acc0 = __builtin_amdgcn_wmma_f32_16x16x32_bf16(false, a, false, bv0, (short)0, acc0, false, false);
                acc1 = __builtin_amdgcn_wmma_f32_16x16x32_bf16(false, a, false, bv1, (short)0, acc1, false, false);
                acc2 = __builtin_amdgcn_wmma_f32_16x16x32_bf16(false, a, false, bv2, (short)0, acc2, false, false);
                acc3 = __builtin_amdgcn_wmma_f32_16x16x32_bf16(false, a, false, bv3, (short)0, acc3, false, false);
            }
            // elementwise LSTM cell: i/f/g/o share lane layout -> pure register math
            const int j = wave * 32 + q * 16 + ln;
#pragma unroll
            for (int r = 0; r < 8; ++r) {
                const int m  = lhalf * 8 + r;            // D layout: M = r + 8*(lane>=16)
                const float xv = x_lds[m][t];
                const float gi = acc0[r] + xv * wih_r[q][0] + bias_r[q][0];
                const float gf = acc1[r] + xv * wih_r[q][1] + bias_r[q][1];
                const float gg = acc2[r] + xv * wih_r[q][2] + bias_r[q][2];
                const float go = acc3[r] + xv * wih_r[q][3] + bias_r[q][3];
                const float cn = sigf(gf) * c_reg[q][r] + sigf(gi) * tanh_fast(gg);
                const float hn = sigf(go) * tanh_fast(cn);
                c_reg[q][r] = cn;
                h_lds[nxt][m][j] = f2bf(hn);
                if (t == T_STEPS - 1)
                    h_out[(size_t)(row0 + m) * HID + j] = hn;
            }
        }
        __syncthreads();     // all writes to nxt done; all reads of cur done
        cur ^= 1;
    }
}

// ---------------- tiny MLP head: 512 -> 256 relu -> 128 relu -> 1 -----------------
__global__ __launch_bounds__(256) void fc_head_kernel(
    const float* __restrict__ h,    // (B, 512)
    const float* __restrict__ W0, const float* __restrict__ b0,
    const float* __restrict__ W1, const float* __restrict__ b1,
    const float* __restrict__ W2, const float* __restrict__ b2,
    float* __restrict__ out)        // (B, 1)
{
    __shared__ float sh[HID];
    __shared__ float s0[256];
    __shared__ float s1[128];
    const int b = blockIdx.x, tid = threadIdx.x;

    for (int i = tid; i < HID; i += 256) sh[i] = h[(size_t)b * HID + i];
    __syncthreads();

    {   // layer 0: 256 neurons, one per thread
        float acc = b0[tid];
        const float* w = W0 + (size_t)tid * HID;
        for (int k = 0; k < HID; ++k) acc += sh[k] * w[k];
        s0[tid] = fmaxf(acc, 0.0f);
    }
    __syncthreads();
    if (tid < 128) {   // layer 1 (+ fold layer-2 weight in)
        float acc = b1[tid];
        const float* w = W1 + (size_t)tid * 256;
        for (int k = 0; k < 256; ++k) acc += s0[k] * w[k];
        s1[tid] = fmaxf(acc, 0.0f) * W2[tid];
    }
    __syncthreads();
    for (int s = 64; s > 0; s >>= 1) {          // reduce 128 -> 1
        if (tid < s) s1[tid] += s1[tid + s];
        __syncthreads();
    }
    if (tid == 0) out[b] = s1[0] + b2[0];
}

// ---------------- launcher ----------------
extern "C" void kernel_launch(void* const* d_in, const int* in_sizes, int n_in,
                              void* d_out, int out_size, void* d_ws, size_t ws_size,
                              hipStream_t stream) {
    (void)in_sizes; (void)n_in; (void)out_size; (void)ws_size;
    const float* x    = (const float*)d_in[0];
    const float* W_ih = (const float*)d_in[1];   // (4H,1) == w_ih_vec
    const float* W_hh = (const float*)d_in[2];
    const float* b_ih = (const float*)d_in[3];
    const float* b_hh = (const float*)d_in[4];
    const float* W0   = (const float*)d_in[5];
    const float* b0   = (const float*)d_in[6];
    const float* W1   = (const float*)d_in[7];
    const float* b1   = (const float*)d_in[8];
    const float* W2   = (const float*)d_in[9];
    const float* b2   = (const float*)d_in[10];
    float* out = (float*)d_out;

    __bf16* Wb      = (__bf16*)d_ws;                                   // 2 MB
    float*  h_final = (float*)((char*)d_ws + (size_t)G4 * HID * 2);    // +1 MB

    const int nW = G4 * HID;
    convert_w_kernel<<<(nW + 255) / 256, 256, 0, stream>>>(W_hh, Wb, nW);
    lstm_scan_kernel<<<BATCH / ROWS, BLOCK, 0, stream>>>(x, W_ih, b_ih, b_hh, Wb, h_final);
    fc_head_kernel<<<BATCH, 256, 0, stream>>>(h_final, W0, b0, W1, b1, W2, b2, out);
}
